// PPNWithUNet_91036126806970
// MI455X (gfx1250) — matrix operand order repaired and checked by reading, not compiled, not run on hardware
//
#include <hip/hip_runtime.h>

typedef __attribute__((ext_vector_type(16))) _Float16 v16h;
typedef __attribute__((ext_vector_type(8)))  float    v8f;

#define MB (1ull<<20)
// workspace layout (bytes)
#define OFF_P0   (0*MB)      // padded f16 input   (18 MB)
#define OFF_P1   (18*MB)     // padded f16 hidden  (18 MB)
#define OFF_Y1   (36*MB)     // conv1 raw f32      (16 MB)  [FEAT spans Y1+Y2]
#define OFF_Y2   (52*MB)     // conv2 raw f32 / upsample raw (16 MB)
#define OFF_IDN  (68*MB)     // residual raw f32 / H1 (16 MB)
#define OFF_X    (84*MB)     // current activation f32 (16 MB)
#define OFF_SK0  (100*MB)
#define OFF_SK1  (116*MB)
#define OFF_SK2  (124*MB)
#define OFF_SK3  (128*MB)
#define OFF_WP   (130*MB)    // packed weights (20 MB)
#define OFF_BN   (150*MB)    // per-channel scale/shift (8 KB)

// ---------------- weight pre-swizzle into WMMA B-fragment layout ----------------
// dst layout: [tap][kb][nb][lane 0..31][elem 0..15] halves
// B 32x16 f16 layout: lane l: n = l%16, h = l/16 ; element e -> K = kb*32 + h*16 + e
// KB, NB are powers of two -> shift/mask decomposition
__global__ void pack_w_kernel(const float* __restrict__ src, _Float16* __restrict__ dst,
                              int ntaps, int Cin, int Cout, int KB, int NB) {
  long total = ((long)ntaps * KB * NB) << 9;
  long idx = (long)blockIdx.x * blockDim.x + threadIdx.x;
  if (idx >= total) return;
  const int nbS = __builtin_ctz(NB);
  const int kbS = __builtin_ctz(KB);
  int e    = idx & 15;
  int lane = (int)((idx >> 4) & 31);
  long rest = idx >> 9;
  int nb = (int)(rest & (NB - 1)); rest >>= nbS;
  int kb = (int)(rest & (KB - 1)); rest >>= kbS;
  int t  = (int)rest;
  int n = nb * 16 + (lane & 15);
  int k = kb * 32 + (lane >> 4) * 16 + e;
  float v = (k < Cin && n < Cout) ? src[((long)t * Cin + k) * Cout + n] : 0.f;
  dst[idx] = (_Float16)v;
}

// ---------------- zero-padded f16 staging (with channel offset for concat) ----------------
// grid: (ceil((W+2)<<chS / 256), H+2) ; covers channels [chOff, chOff + (1<<chS))
__global__ void pad_f16_kernel(const float* __restrict__ src, _Float16* __restrict__ dst,
                               int W, int srcC, int chS, int dstC, int chOff) {
  const int r = blockIdx.y;
  const int H = gridDim.y - 2;
  long tot = (long)(W + 2) << chS;
  long idx = (long)blockIdx.x * blockDim.x + threadIdx.x;
  if (idx >= tot) return;
  int ch = (int)(idx & ((1 << chS) - 1));
  int c  = (int)(idx >> chS);
  float v = 0.f;
  if (r >= 1 && r <= H && c >= 1 && c <= W && ch < srcC)
    v = src[(((long)(r - 1)) * W + (c - 1)) * srcC + ch];
  dst[((long)r * (W + 2) + c) * dstC + chOff + ch] = (_Float16)v;
}

// ---------------- BN batch stats -> per-channel scale/shift ----------------
__global__ void bn_stats_kernel(const float* __restrict__ x, const float* __restrict__ g,
                                const float* __restrict__ b, float* __restrict__ ss,
                                int HW, int C) {
  int ch = blockIdx.x;
  float s = 0.f, s2 = 0.f;
  for (int i = threadIdx.x; i < HW; i += blockDim.x) {
    float v = x[(long)i * C + ch];
    s += v; s2 += v * v;
  }
  __shared__ float sh[512];
  sh[threadIdx.x] = s; sh[256 + threadIdx.x] = s2;
  __syncthreads();
  for (int o = 128; o > 0; o >>= 1) {
    if ((int)threadIdx.x < o) {
      sh[threadIdx.x] += sh[threadIdx.x + o];
      sh[256 + threadIdx.x] += sh[256 + threadIdx.x + o];
    }
    __syncthreads();
  }
  if (threadIdx.x == 0) {
    float m = sh[0] / (float)HW;
    float var = sh[256] / (float)HW - m * m;
    float sc = g[ch] * rsqrtf(var + 1e-5f);
    ss[ch] = sc;
    ss[C + ch] = b[ch] - m * sc;
  }
}

// BN + ReLU fused with pad-to-f16 (C == dstC == 1<<chS here)
// grid: (ceil((W+2)<<chS / 256), H+2)
__global__ void bn_pad_f16_kernel(const float* __restrict__ src, const float* __restrict__ ss,
                                  _Float16* __restrict__ dst, int W, int chS) {
  const int r = blockIdx.y;
  const int H = gridDim.y - 2;
  long tot = (long)(W + 2) << chS;
  long idx = (long)blockIdx.x * blockDim.x + threadIdx.x;
  if (idx >= tot) return;
  int ch = (int)(idx & ((1 << chS) - 1));
  int c  = (int)(idx >> chS);
  float v = 0.f;
  if (r >= 1 && r <= H && c >= 1 && c <= W) {
    float x = src[((((long)(r - 1)) * W + (c - 1)) << chS) + ch];
    v = fmaxf(x * ss[ch] + ss[(1 << chS) + ch], 0.f);
  }
  dst[(((long)r * (W + 2) + c) << chS) + ch] = (_Float16)v;
}

// out = relu(bn(y2) + idn)  (C power of two)
__global__ void bn_add_relu_kernel(const float* __restrict__ y, const float* __restrict__ idn,
                                   const float* __restrict__ ss, float* __restrict__ out,
                                   long n, int C) {
  long i = (long)blockIdx.x * blockDim.x + threadIdx.x;
  if (i >= n) return;
  int ch = (int)(i & (C - 1));
  out[i] = fmaxf(y[i] * ss[ch] + ss[C + ch] + idn[i], 0.f);
}

// C, W powers of two
__global__ void maxpool_kernel(const float* __restrict__ x, float* __restrict__ y,
                               int H, int W, int C) {
  long n = ((long)(H / 2) * (W / 2)) * C;
  long i = (long)blockIdx.x * blockDim.x + threadIdx.x;
  if (i >= n) return;
  const int cS = __builtin_ctz(C);
  const int wS = __builtin_ctz(W / 2);
  int ch = (int)(i & (C - 1));
  long p = i >> cS;
  int c = (int)(p & ((W / 2) - 1));
  int r = (int)(p >> wS);
  const float* b = x + ((((long)(2 * r)) * W + 2 * c) << cS) + ch;
  long rs = (long)W << cS;
  y[i] = fmaxf(fmaxf(b[0], b[C]), fmaxf(b[rs], b[rs + C]));
}

__global__ void bias_relu_kernel(float* __restrict__ x, const float* __restrict__ b,
                                 long n, int C) {
  long i = (long)blockIdx.x * blockDim.x + threadIdx.x;
  if (i >= n) return;
  x[i] = fmaxf(x[i] + b[(int)(i & (C - 1))], 0.f);
}

// ---------------- implicit-GEMM conv via WMMA ----------------
// inPad: (H+2, W+2, Cpad) f16, zero border/channel-pad
// wpack: [ntaps][KB][NB][32][16] halves (pre-swizzled B fragments)
// Each wave: MACC*16 consecutive pixels x NACC*16 out channels (8 f32 accs).
// Exact grid (tilesM % 8 == 0, Cout % 64 == 0 for every layer): no guard, no
// divisions (W power of two -> shifts), EXEC provably all-ones around WMMA.
// Store mapping generalized for transposed conv: orow=row*rs+ro, ocol=col*cs+co.
template<int MACC, int NACC>
__global__ __launch_bounds__(256) void conv_wmma_kernel(
    const _Float16* __restrict__ inPad, const _Float16* __restrict__ wpack,
    float* __restrict__ out, int Wlog2, int Cpad, int Cout, int ntaps,
    int oW, int rs, int ro, int cs, int co) {
  const int lane = threadIdx.x & 31;
  const int wv   = threadIdx.x >> 5;
  const int tm   = blockIdx.x * 8 + wv;   // M super-tile (MACC*16 pixels)
  const int tn   = blockIdx.y;            // N super-tile (NACC*16 channels)
  const int m = lane & 15;
  const int h = lane >> 4;
  const int W  = 1 << Wlog2;
  const int Wp = W + 2;
  const int KB = Cpad >> 5;
  const int NB = Cout >> 4;
  const int cpS = __builtin_ctz(Cpad);    // uniform scalar shifts
  const int coS = __builtin_ctz(Cout);

  int row[MACC], col[MACC];
#pragma unroll
  for (int i = 0; i < MACC; i++) {
    int pix = (tm * MACC + i) * 16;       // each 16-tile stays within one row (W%16==0)
    row[i] = pix >> Wlog2;
    col[i] = pix & (W - 1);
  }

  const uint4* inV = (const uint4*)inPad;  // 8 halves per uint4, 16B aligned
  const uint4* wV  = (const uint4*)wpack;

  v8f acc[MACC][NACC];
#pragma unroll
  for (int i = 0; i < MACC; i++)
#pragma unroll
    for (int j = 0; j < NACC; j++)
#pragma unroll
      for (int r = 0; r < 8; r++) acc[i][j][r] = 0.f;

  for (int t = 0; t < ntaps; t++) {
    const int pdy = (ntaps == 9) ? (t / 3) : 1;
    const int pdx = (ntaps == 9) ? (t % 3) : 1;
    for (int kb = 0; kb < KB; kb++) {
      if (kb + 1 < KB)
        __builtin_prefetch(
            (const void*)((const unsigned*)wpack +
                          ((((long)t * KB + kb + 1) * NB + tn * NACC) * 32 + lane) * 8),
            0, 0);
      // A fragments: u[0..7] = halves [h*16, h*16+16)
      union { uint4 q[2]; v16h f; } A[MACC];
#pragma unroll
      for (int i = 0; i < MACC; i++) {
        long base = (((long)(row[i] + pdy) * Wp + (col[i] + m + pdx)) << cpS) +
                    kb * 32 + h * 16;     // half index, multiple of 8 -> 16B aligned
        A[i].q[0] = inV[base >> 3];
        A[i].q[1] = inV[(base + 8) >> 3];
      }
#pragma unroll
      for (int j = 0; j < NACC; j++) {
        union { uint4 q[2]; v16h f; } B;
        long wb = ((((long)t * KB + kb) * NB + (tn * NACC + j)) * 32 + lane) * 16; // halves
        B.q[0] = wV[wb >> 3];
        B.q[1] = wV[(wb >> 3) + 1];
#pragma unroll
        for (int i = 0; i < MACC; i++)
          acc[i][j] = __builtin_amdgcn_wmma_f32_16x16x32_f16(
              false, A[i].f, false, B.f, (short)0, acc[i][j], false, false);
      }
    }
  }
  // C/D layout: lane l: n = l%16 ; VGPR r: m = r + 8*(l/16)
#pragma unroll
  for (int i = 0; i < MACC; i++) {
    int orow = row[i] * rs + ro;
#pragma unroll
    for (int j = 0; j < NACC; j++) {
      int n = (tn * NACC + j) * 16 + (lane & 15);
#pragma unroll
      for (int r = 0; r < 8; r++) {
        int mm = r + 8 * (lane >> 4);
        int ocol = (col[i] + mm) * cs + co;
        out[((((long)orow * oW + ocol)) << coS) + n] = acc[i][j][r];
      }
    }
  }
}

// ---------------- per-pixel head: fc2(+bias), sigmoid(fcc), mask, pred ----------------
__global__ void head_kernel(const float* __restrict__ h1, const float* __restrict__ feat,
                            const float* __restrict__ fc2w, const float* __restrict__ fc2b,
                            const float* __restrict__ fccw, const float* __restrict__ fccb,
                            float* __restrict__ out, int HW) {
  int p = blockIdx.x * blockDim.x + threadIdx.x;
  if (p >= HW) return;
  float r0 = fc2b[0], r1 = fc2b[1], r2 = fc2b[2], r3 = fc2b[3];
  const float4* h4 = (const float4*)(h1 + ((long)p << 6));
  for (int c4 = 0; c4 < 16; c4++) {
    float4 hv = h4[c4];
    const float* w = fc2w + c4 * 16;
    r0 += hv.x * w[0]  + hv.y * w[4]  + hv.z * w[8]  + hv.w * w[12];
    r1 += hv.x * w[1]  + hv.y * w[5]  + hv.z * w[9]  + hv.w * w[13];
    r2 += hv.x * w[2]  + hv.y * w[6]  + hv.z * w[10] + hv.w * w[14];
    r3 += hv.x * w[3]  + hv.y * w[7]  + hv.z * w[11] + hv.w * w[15];
  }
  float z = fccb[0];
  const float4* f4 = (const float4*)(feat + ((long)p << 7));
  for (int c4 = 0; c4 < 32; c4++) {
    float4 fv = f4[c4];
    const float* w = fccw + c4 * 4;
    z += fv.x * w[0] + fv.y * w[1] + fv.z * w[2] + fv.w * w[3];
  }
  float cls = 1.f / (1.f + expf(-z));
  float mask = (cls > 0.5f) ? 1.f : 0.f;
  float* reg = out + ((long)p << 2);
  reg[0] = r0; reg[1] = r1; reg[2] = r2; reg[3] = r3;
  out[262144 + p] = cls;
  float* pr = out + 393216 + ((long)p << 2);
  pr[0] = r0 * mask; pr[1] = r1 * mask; pr[2] = r2 * mask; pr[3] = r3 * mask;
}

// 3x3 stride-1 SAME conv_transpose (4 -> 1), kernel spatially flipped; W=256
__global__ void dec_kernel(const float* __restrict__ pred, const float* __restrict__ w,
                           float* __restrict__ out, int H, int W) {
  int i = blockIdx.x * blockDim.x + threadIdx.x;
  if (i >= H * W) return;
  const int wS = __builtin_ctz(W);
  int c0 = i & (W - 1), r0 = i >> wS;
  float s = 0.f;
  for (int a = 0; a < 3; a++)
    for (int b = 0; b < 3; b++) {
      int r = r0 + 1 - a, c = c0 + 1 - b;
      if (r >= 0 && r < H && c >= 0 && c < W) {
        float4 p = *(const float4*)(pred + ((((long)r << wS) + c) << 2));
        const float* ww = w + (a * 3 + b) * 4;
        s += p.x * ww[0] + p.y * ww[1] + p.z * ww[2] + p.w * ww[3];
      }
    }
  out[i] = s;
}

// ---------------- host orchestration ----------------
struct DCP { const float *b1, *b2, *g1, *g2, *res, *w1, *w2; };

static inline void launch_pack(const float* w, _Float16* dst, int ntaps, int Cin, int Cout,
                               hipStream_t s) {
  int KB = (Cin + 31) / 32, NB = (Cout + 15) / 16;
  long total = ((long)ntaps * KB * NB) << 9;
  pack_w_kernel<<<(unsigned)((total + 255) / 256), 256, 0, s>>>(w, dst, ntaps, Cin, Cout, KB, NB);
}

static inline void launch_conv(const _Float16* inPad, const _Float16* wp, float* out,
                               int H, int W, int Cpad, int Cout, int ntaps,
                               int oW, int rs, int ro, int cs, int co, hipStream_t s) {
  // every layer here: H*W/32 % 8 == 0 and Cout % 64 == 0 -> exact grid
  int tilesM = (H * W) / 32;
  int tilesN = Cout / 64;
  dim3 grid((unsigned)(tilesM / 8), (unsigned)tilesN);
  conv_wmma_kernel<2, 4><<<grid, 256, 0, s>>>(inPad, wp, out, __builtin_ctz(W), Cpad, Cout,
                                              ntaps, oW, rs, ro, cs, co);
}

static inline void launch_pad(const float* src, _Float16* dst, int H, int W, int srcC,
                              int dstC, int chOff, int nCh, hipStream_t s) {
  int chS = __builtin_ctz(nCh);
  long rowTot = (long)(W + 2) << chS;
  dim3 grid((unsigned)((rowTot + 255) / 256), (unsigned)(H + 2));
  pad_f16_kernel<<<grid, 256, 0, s>>>(src, dst, W, srcC, chS, dstC, chOff);
}

struct WS {
  _Float16 *P0, *P1, *WP;
  float *Y1, *Y2, *IDN, *X, *BN;
};

// input already staged as padded f16 in ws.P0 (Cpad = roundup(Cin,32))
static void double_conv(const WS& ws, const DCP& p, int Cin, int Cout, int H, int W,
                        float* out, hipStream_t s) {
  int CinP = ((Cin + 31) / 32) * 32;
  // conv1
  launch_pack(p.w1, ws.WP, 9, Cin, Cout, s);
  launch_conv(ws.P0, ws.WP, ws.Y1, H, W, CinP, Cout, 9, W, 1, 0, 1, 0, s);
  bn_stats_kernel<<<Cout, 256, 0, s>>>(ws.Y1, p.g1, p.b1, ws.BN, H * W, Cout);
  {
    int chS = __builtin_ctz(Cout);
    long rowTot = (long)(W + 2) << chS;
    dim3 g((unsigned)((rowTot + 255) / 256), (unsigned)(H + 2));
    bn_pad_f16_kernel<<<g, 256, 0, s>>>(ws.Y1, ws.BN, ws.P1, W, chS);
  }
  // conv2
  launch_pack(p.w2, ws.WP, 9, Cout, Cout, s);
  launch_conv(ws.P1, ws.WP, ws.Y2, H, W, Cout, Cout, 9, W, 1, 0, 1, 0, s);
  // residual 1x1 (reads center tap of P0)
  launch_pack(p.res, ws.WP, 1, Cin, Cout, s);
  launch_conv(ws.P0, ws.WP, ws.IDN, H, W, CinP, Cout, 1, W, 1, 0, 1, 0, s);
  bn_stats_kernel<<<Cout, 256, 0, s>>>(ws.Y2, p.g2, p.b2, ws.BN, H * W, Cout);
  long n = (long)H * W * Cout;
  bn_add_relu_kernel<<<(unsigned)((n + 255) / 256), 256, 0, s>>>(ws.Y2, ws.IDN, ws.BN, out, n, Cout);
}

extern "C" void kernel_launch(void* const* d_in, const int* in_sizes, int n_in,
                              void* d_out, int out_size, void* d_ws, size_t ws_size,
                              hipStream_t stream) {
  (void)in_sizes; (void)n_in; (void)out_size; (void)ws_size;
  char* wsb = (char*)d_ws;
  WS ws;
  ws.P0  = (_Float16*)(wsb + OFF_P0);
  ws.P1  = (_Float16*)(wsb + OFF_P1);
  ws.WP  = (_Float16*)(wsb + OFF_WP);
  ws.Y1  = (float*)(wsb + OFF_Y1);
  ws.Y2  = (float*)(wsb + OFF_Y2);
  ws.IDN = (float*)(wsb + OFF_IDN);
  ws.X   = (float*)(wsb + OFF_X);
  ws.BN  = (float*)(wsb + OFF_BN);
  float* skips[4] = { (float*)(wsb + OFF_SK0), (float*)(wsb + OFF_SK1),
                      (float*)(wsb + OFF_SK2), (float*)(wsb + OFF_SK3) };

  auto F = [&](int i) { return (const float*)d_in[i]; };
  const float* features = F(0);
  // params pytree order (dicts sorted by key):
  DCP bott = { F(1), F(2), F(3), F(4), F(5), F(6), F(7) };
  const float* cls_w = F(8);
  const float* dec_w = F(9);
  DCP down[4];
  for (int i = 0; i < 4; i++) {
    int b = 10 + 7 * i;
    down[i] = { F(b), F(b + 1), F(b + 2), F(b + 3), F(b + 4), F(b + 5), F(b + 6) };
  }
  const float* fc1_b = F(38); const float* fc1_w = F(39);
  const float* fc2_b = F(40); const float* fc2_w = F(41);
  const float* fcc_b = F(42); const float* fcc_w = F(43);
  DCP updc[4]; const float* uptw[4];
  for (int i = 0; i < 4; i++) {
    int b = 44 + 8 * i;
    updc[i] = { F(b), F(b + 1), F(b + 2), F(b + 3), F(b + 4), F(b + 5), F(b + 6) };
    uptw[i] = F(b + 7);
  }

  const int FILTS[4] = { 64, 128, 256, 512 };
  int H = 256, W = 256;

  // stage input: 1 channel -> padded f16 (Cpad=32)
  launch_pad(features, ws.P0, H, W, 1, 32, 0, 32, stream);

  // encoder
  int Cin = 1;
  for (int i = 0; i < 4; i++) {
    int f = FILTS[i];
    double_conv(ws, down[i], Cin, f, H, W, skips[i], stream);
    long n = (long)(H / 2) * (W / 2) * f;
    maxpool_kernel<<<(unsigned)((n + 255) / 256), 256, 0, stream>>>(skips[i], ws.X, H, W, f);
    H /= 2; W /= 2;
    launch_pad(ws.X, ws.P0, H, W, f, f, 0, f, stream);
    Cin = f;
  }

  // bottleneck (16x16, 512 -> 1024) -> ws.X
  double_conv(ws, bott, 512, 1024, H, W, ws.X, stream);
  int C = 1024;

  // decoder
  for (int i = 0; i < 4; i++) {
    int f = FILTS[3 - i];            // 512,256,128,64 ; C == 2f
    launch_pad(ws.X, ws.P0, H, W, C, C, 0, C, stream);
    // 2x2 stride-2 conv_transpose = 4 one-tap GEMMs with scattered stores
    launch_pack(uptw[i], ws.WP, 4, C, f, stream);
    int KB = (C + 31) / 32, NB = (f + 15) / 16;
    float* UP = ws.Y2;
    for (int di = 0; di < 2; di++)
      for (int dj = 0; dj < 2; dj++) {
        const _Float16* wslice = ws.WP + (((long)(di * 2 + dj) * KB * NB) << 9);
        launch_conv(ws.P0, wslice, UP, H, W, C, f, 1, 2 * W, 2, di, 2, dj, stream);
      }
    H *= 2; W *= 2;
    // concat(skip, up) directly into padded f16 buffer
    launch_pad(skips[3 - i], ws.P0, H, W, f, 2 * f, 0, f, stream);
    launch_pad(UP,           ws.P0, H, W, f, 2 * f, f, f, stream);
    double_conv(ws, updc[i], 2 * f, f, H, W, ws.X, stream);
    C = f;
  }

  // heads: H=W=256, C=64
  float* FEAT = ws.Y1;               // 32 MB spans Y1+Y2 (free now)
  float* H1   = ws.IDN;
  launch_pad(ws.X, ws.P0, H, W, 64, 64, 0, 64, stream);
  launch_pack(cls_w, ws.WP, 1, 64, 128, stream);
  launch_conv(ws.P0, ws.WP, FEAT, H, W, 64, 128, 1, W, 1, 0, 1, 0, stream);
  launch_pad(FEAT, ws.P1, H, W, 128, 128, 0, 128, stream);
  launch_pack(fc1_w, ws.WP, 1, 128, 64, stream);
  launch_conv(ws.P1, ws.WP, H1, H, W, 128, 64, 1, W, 1, 0, 1, 0, stream);
  long n1 = (long)H * W * 64;
  bias_relu_kernel<<<(unsigned)((n1 + 255) / 256), 256, 0, stream>>>(H1, fc1_b, n1, 64);

  float* out = (float*)d_out;
  head_kernel<<<(65536 + 255) / 256, 256, 0, stream>>>(H1, FEAT, fc2_w, fc2_b, fcc_w, fcc_b,
                                                       out, 65536);
  dec_kernel<<<(65536 + 255) / 256, 256, 0, stream>>>(out + 393216, dec_w, out + 327680, 256, 256);
}
